// ElementwiseTensorProducts_9062380994711
// MI455X (gfx1250) — matrix operand order repaired and checked by reading, not compiled, not run on hardware
//
#include <hip/hip_runtime.h>

// ---------------------------------------------------------------------------
// MI455X (gfx1250) fused equivariant tensor-product block, v3.1.
// - All GEMMs via v_wmma_f32_16x16x32_bf16 (wave32 WMMA).
// - Rank-tile-at-a-time projection phase keeps live VGPRs < 256 (no spills).
// - f32->bf16 via __builtin_convertvector so the backend can pick packed cvt.
// Roofline: 268 MB activation traffic @ 23.3 TB/s ~= 11.5 us floor; 20.4 GFLOP
// of bf16 WMMA work sits under that -> memory bound.
// ---------------------------------------------------------------------------

typedef __attribute__((ext_vector_type(16))) __bf16 v16bf;
typedef __attribute__((ext_vector_type(4)))  __bf16 v4bf;
typedef __attribute__((ext_vector_type(2)))  __bf16 v2bf;
typedef __attribute__((ext_vector_type(8)))  float  v8f;
typedef __attribute__((ext_vector_type(4)))  float  vf4;

union FragU {
  v16bf v;
  v4bf  b4[4];
  uint4 q[2];
  unsigned int u[8];
  unsigned short h[16];
};

__device__ __forceinline__ unsigned short bf1(float x) {
  union { __bf16 b; unsigned short s; } c;
  c.b = (__bf16)x;
  return c.s;
}

__device__ __forceinline__ unsigned int pk2(float a, float b) {
  union { v2bf v; unsigned int u; } c;
  c.v[0] = (__bf16)a;
  c.v[1] = (__bf16)b;
  return c.u;
}

__device__ __forceinline__ v8f wmma_bf16(v16bf a, v16bf b, v8f c) {
  return __builtin_amdgcn_wmma_f32_16x16x32_bf16(false, a, false, b,
                                                 (short)0, c, false, false);
}

__device__ __forceinline__ v8f splat8(float x) {
  v8f v;
#pragma unroll
  for (int j = 0; j < 8; ++j) v[j] = x;
  return v;
}

// A-fragment (16x32 bf16, ISA layout): lane(0-15)=row M, holds K=k0..k0+7 in
// v0-3 and K=k0+16..k0+23 in v4-7 (k0 has +8 for lanes 16-31).
__device__ __forceinline__ v16bf load_a_f32(const float* __restrict__ row, int k0) {
  FragU f;
  f.b4[0] = __builtin_convertvector(*(const vf4*)(row + k0),      v4bf);
  f.b4[1] = __builtin_convertvector(*(const vf4*)(row + k0 + 4),  v4bf);
  f.b4[2] = __builtin_convertvector(*(const vf4*)(row + k0 + 16), v4bf);
  f.b4[3] = __builtin_convertvector(*(const vf4*)(row + k0 + 20), v4bf);
  return f.v;
}

// B-fragment from pre-swizzled LDS: 32 contiguous bytes per lane (2x b128).
__device__ __forceinline__ v16bf load_b_frag(const unsigned short* s_w, int frag, int lane) {
  FragU f;
  const unsigned short* p = s_w + frag * 512 + lane * 16;
  f.q[0] = *(const uint4*)(p);
  f.q[1] = *(const uint4*)(p + 8);
  return f.v;
}

// A-fragment from the per-wave staging tile (bf16, row pitch 200 halves).
__device__ __forceinline__ v16bf load_stage_a(const unsigned short* st, int row16,
                                              int kt, int ln, int hi) {
  FragU f;
  const unsigned short* p = st + (row16 * 16 + ln) * 200 + kt * 32 + hi * 8;
  f.q[0] = *(const uint4*)(p);
  f.q[1] = *(const uint4*)(p + 16);
  return f.v;
}

#define CHAN 128
#define SW_HALVES 73728          // 144 fragments * 512 halves (144 KB)
#define STAGE_PITCH 200          // halves per staged row (192 cols + pad)
#define STAGE_PER_WAVE (48 * STAGE_PITCH)

__global__ __launch_bounds__(256) void etp_kernel(
    const float* __restrict__ z0,  const float* __restrict__ z1,
    const float* __restrict__ W0l, const float* __restrict__ b0l,
    const float* __restrict__ W0r, const float* __restrict__ b0r,
    const float* __restrict__ W1l, const float* __restrict__ W1r,
    const float* __restrict__ W0o, const float* __restrict__ b0o,
    const float* __restrict__ W1o,
    float* __restrict__ out0, float* __restrict__ out1)
{
  extern __shared__ unsigned short smem[];
  unsigned short* s_w = smem;                      // swizzled weights
  unsigned short* s_stage = smem + SW_HALVES;      // per-wave staging tiles

  const int tid  = threadIdx.x;
  const int wave = tid >> 5;
  const int lane = tid & 31;
  const int ln   = lane & 15;
  const int hi   = lane >> 4;

  // -------- stage all weights into LDS in B-fragment layout ---------------
  // frag ids: W0l:0-15  W0r:16-31  W1l:32-47  W1r:48-63  W0o:64-95  W1o:96-143
  for (int it = 0; it < 18; ++it) {
    int idx = it * 256 + tid;                  // 0..4607 fragment-lanes
    int f   = idx >> 5;
    int ls  = idx & 31;
    int lsn = ls & 15, lsh = ls >> 4;
    const float* src;
    int nt, kt, rowlen;
    if (f < 64) {
      int m = f >> 4, r = f & 15;
      nt = r >> 2; kt = r & 3; rowlen = CHAN;
      src = (m == 0) ? W0l : (m == 1) ? W0r : (m == 2) ? W1l : W1r;
    } else if (f < 96) {
      int r = f - 64;
      nt = r >> 2; kt = r & 3; rowlen = 128; src = W0o;
    } else {
      int r = f - 96;
      nt = r / 6; kt = r % 6; rowlen = 192; src = W1o;
    }
    const float* p = src + (nt * 16 + lsn) * rowlen + kt * 32 + lsh * 8;
    FragU f8;
    f8.b4[0] = __builtin_convertvector(*(const vf4*)(p),      v4bf);
    f8.b4[1] = __builtin_convertvector(*(const vf4*)(p + 4),  v4bf);
    f8.b4[2] = __builtin_convertvector(*(const vf4*)(p + 16), v4bf);
    f8.b4[3] = __builtin_convertvector(*(const vf4*)(p + 20), v4bf);
    uint4* dst = (uint4*)(s_w + f * 512 + ls * 16);
    dst[0] = f8.q[0];
    dst[1] = f8.q[1];
  }
  __syncthreads();

  const long tokenBase = (long)blockIdx.x * 128 + wave * 16;
  const int  k0a = hi ? 8 : 0;
  unsigned short* st = s_stage + wave * STAGE_PER_WAVE;

  // per-lane biases (depend only on column index ln within each n-tile)
  float blv[4], brv[4], bo[8];
#pragma unroll
  for (int nt = 0; nt < 4; ++nt) { blv[nt] = b0l[nt * 16 + ln]; brv[nt] = b0r[nt * 16 + ln]; }
#pragma unroll
  for (int nt = 0; nt < 8; ++nt) { bo[nt] = b0o[nt * 16 + ln]; }

  // -------- resident A fragments: 16 tokens x (z0 + 3 z1 comps) ------------
  v16bf a0[4], a1[3][4];
  {
    const float* z0row = z0 + (tokenBase + ln) * (long)CHAN;
#pragma unroll
    for (int kt = 0; kt < 4; ++kt) a0[kt] = load_a_f32(z0row, kt * 32 + k0a);
#pragma unroll
    for (int i = 0; i < 3; ++i) {
      const float* z1row = z1 + ((tokenBase + ln) * 3 + i) * (long)CHAN;
#pragma unroll
      for (int kt = 0; kt < 4; ++kt) a1[i][kt] = load_a_f32(z1row, kt * 32 + k0a);
    }
  }

  // p00 / p110 held packed (bf16 pairs) until the o0 phase
  unsigned int p0h[4][4], p1h[4][4];

  // -------- projections + elementwise, one rank n-tile at a time -----------
#pragma unroll
  for (int nt = 0; nt < 4; ++nt) {
    v8f zl = splat8(blv[nt]);
    v8f zr = splat8(brv[nt]);
    v8f l[3], r[3];
#pragma unroll
    for (int i = 0; i < 3; ++i) { l[i] = splat8(0.f); r[i] = splat8(0.f); }

#pragma unroll
    for (int kt = 0; kt < 4; ++kt) {
      zl = wmma_bf16(a0[kt], load_b_frag(s_w, 0  + nt * 4 + kt, lane), zl);
      zr = wmma_bf16(a0[kt], load_b_frag(s_w, 16 + nt * 4 + kt, lane), zr);
      v16bf bL = load_b_frag(s_w, 32 + nt * 4 + kt, lane);   // shared by 3 comps
#pragma unroll
      for (int i = 0; i < 3; ++i) l[i] = wmma_bf16(a1[i][kt], bL, l[i]);
      v16bf bR = load_b_frag(s_w, 48 + nt * 4 + kt, lane);
#pragma unroll
      for (int i = 0; i < 3; ++i) r[i] = wmma_bf16(a1[i][kt], bR, r[i]);
    }

    // p00 / p110 -> packed registers
#pragma unroll
    for (int jj = 0; jj < 4; ++jj) {
      float pa = zl[2 * jj] * zr[2 * jj];
      float pb = zl[2 * jj + 1] * zr[2 * jj + 1];
      p0h[nt][jj] = pk2(pa, pb);
      float sa = l[0][2 * jj] * r[0][2 * jj]
               + l[1][2 * jj] * r[1][2 * jj]
               + l[2][2 * jj] * r[2][2 * jj];
      float sb = l[0][2 * jj + 1] * r[0][2 * jj + 1]
               + l[1][2 * jj + 1] * r[1][2 * jj + 1]
               + l[2][2 * jj + 1] * r[2][2 * jj + 1];
      p1h[nt][jj] = pk2(sa, sb);
    }

    // q011 | q101 | cross -> staged z1_o tile (48 rows)
    const int kcol = nt * 16 + ln;
#pragma unroll
    for (int i = 0; i < 3; ++i) {
      const int i1 = (i + 1) % 3, i2 = (i + 2) % 3;
#pragma unroll
      for (int j = 0; j < 8; ++j) {
        int rowoff = (i * 16 + j + hi * 8) * STAGE_PITCH;
        st[rowoff + kcol]       = bf1(zl[j] * r[i][j]);
        st[rowoff + 64 + kcol]  = bf1(l[i][j] * zr[j]);
        st[rowoff + 128 + kcol] = bf1(l[i1][j] * r[i2][j] - l[i2][j] * r[i1][j]);
      }
    }
  }
  __syncthreads();

  // -------- out1 GEMM: two 4-column halves; each W1o frag fetched once -----
#pragma unroll
  for (int half = 0; half < 2; ++half) {
    v8f o1[3][4];
#pragma unroll
    for (int i = 0; i < 3; ++i)
#pragma unroll
      for (int n4 = 0; n4 < 4; ++n4) o1[i][n4] = splat8(0.f);

#pragma unroll
    for (int kt = 0; kt < 6; ++kt) {
      v16bf a[3];
#pragma unroll
      for (int i = 0; i < 3; ++i) a[i] = load_stage_a(st, i, kt, ln, hi);
#pragma unroll
      for (int n4 = 0; n4 < 4; ++n4) {
        int nt = half * 4 + n4;
        v16bf b = load_b_frag(s_w, 96 + nt * 6 + kt, lane);
#pragma unroll
        for (int i = 0; i < 3; ++i) o1[i][n4] = wmma_bf16(a[i], b, o1[i][n4]);
      }
    }
#pragma unroll
    for (int i = 0; i < 3; ++i)
#pragma unroll
      for (int n4 = 0; n4 < 4; ++n4)
#pragma unroll
        for (int j = 0; j < 8; ++j) {
          int m = j + hi * 8;
          int c = (half * 4 + n4) * 16 + ln;
          out1[((tokenBase + m) * 3 + i) * (long)CHAN + c] = o1[i][n4][j];
        }
  }
  __syncthreads();   // staging tile about to be overwritten with z0_o

  // -------- restage o0 (p00 | p110) from packed registers ------------------
#pragma unroll
  for (int nt = 0; nt < 4; ++nt) {
    const int kcol = nt * 16 + ln;
#pragma unroll
    for (int jj = 0; jj < 4; ++jj) {
      unsigned int u0 = p0h[nt][jj];
      unsigned int u1 = p1h[nt][jj];
      int r0 = (2 * jj + hi * 8) * STAGE_PITCH;
      int r1 = (2 * jj + 1 + hi * 8) * STAGE_PITCH;
      st[r0 + kcol]      = (unsigned short)u0;
      st[r1 + kcol]      = (unsigned short)(u0 >> 16);
      st[r0 + 64 + kcol] = (unsigned short)u1;
      st[r1 + 64 + kcol] = (unsigned short)(u1 >> 16);
    }
  }
  __syncthreads();

  // -------- out0 GEMM ------------------------------------------------------
  {
    v8f o[8];
#pragma unroll
    for (int nt = 0; nt < 8; ++nt) o[nt] = splat8(bo[nt]);
#pragma unroll
    for (int kt = 0; kt < 4; ++kt) {
      v16bf a = load_stage_a(st, 0, kt, ln, hi);
#pragma unroll
      for (int nt = 0; nt < 8; ++nt)
        o[nt] = wmma_bf16(a, load_b_frag(s_w, 64 + nt * 4 + kt, lane), o[nt]);
    }
#pragma unroll
    for (int nt = 0; nt < 8; ++nt)
#pragma unroll
      for (int j = 0; j < 8; ++j) {
        int m = j + hi * 8;
        out0[(tokenBase + m) * (long)CHAN + nt * 16 + ln] = o[nt][j];
      }
  }
}

extern "C" void kernel_launch(void* const* d_in, const int* in_sizes, int n_in,
                              void* d_out, int out_size, void* d_ws, size_t ws_size,
                              hipStream_t stream) {
  (void)in_sizes; (void)n_in; (void)out_size; (void)d_ws; (void)ws_size;
  const float* z0  = (const float*)d_in[0];
  const float* z1  = (const float*)d_in[1];
  const float* W0l = (const float*)d_in[2];
  const float* b0l = (const float*)d_in[3];
  const float* W0r = (const float*)d_in[4];
  const float* b0r = (const float*)d_in[5];
  const float* W1l = (const float*)d_in[6];
  const float* W1r = (const float*)d_in[7];
  const float* W0o = (const float*)d_in[8];
  const float* b0o = (const float*)d_in[9];
  const float* W1o = (const float*)d_in[10];

  float* out0 = (float*)d_out;
  float* out1 = out0 + (size_t)16 * 4096 * 128;   // out0 is B*N*C floats

  const size_t shmem = (size_t)(SW_HALVES + 8 * STAGE_PER_WAVE) * sizeof(unsigned short);
  dim3 grid(512), block(256);                      // 128 tokens per block
  hipLaunchKernelGGL(etp_kernel, grid, block, shmem, stream,
                     z0, z1, W0l, b0l, W0r, b0r, W1l, W1r, W0o, b0o, W1o,
                     out0, out1);
}
